// MaxMatching_47923245089138
// MI455X (gfx1250) — matrix operand order repaired and checked by reading, not compiled, not run on hardware
//
#include <hip/hip_runtime.h>

typedef __attribute__((ext_vector_type(16))) _Float16 v16h;
typedef __attribute__((ext_vector_type(8)))  _Float16 v8h;
typedef __attribute__((ext_vector_type(4)))  _Float16 v4h;
typedef __attribute__((ext_vector_type(8)))  float    v8f;

#define BB 64
#define LL 200
#define DD 300
#define PP 16

#define KC    64             // K chunk width (2 WMMA k-steps)
#define NKC   5              // ceil(300/64)
#define KP    72             // padded LDS row stride in f16 (144B -> 16 distinct banks)
#define NT    13             // 208 columns = 13 n-tiles
#define NROW  208
#define MROWS 128            // A rows per workgroup (8 m-tiles)

// ---------------- Phase 0: inverse norms rsq[b,l,p] = rsqrt(max(sum_d (s*k)^2, eps)) ----
__global__ __launch_bounds__(256) void mpm_norms(const float* __restrict__ s1,
                                                 const float* __restrict__ s2,
                                                 const float* __restrict__ kern,
                                                 float* __restrict__ rsq1,
                                                 float* __restrict__ rsq2) {
    const int total = BB * LL * PP;
    int idx = blockIdx.x * 256 + threadIdx.x;
    const float* s = s1;
    float* dst = rsq1;
    int t = idx;
    if (idx >= total) { s = s2; dst = rsq2; t = idx - total; }
    if (t >= total) return;

    const int p  = t & (PP - 1);
    const int bl = t >> 4;                       // b*LL + l
    const float4* srow = (const float4*)(s + (size_t)bl * DD);
    const float4* krow = (const float4*)(kern + (size_t)p * DD);
    float acc = 0.f;
#pragma unroll 5
    for (int d4 = 0; d4 < DD / 4; ++d4) {
        float4 sv = srow[d4];
        float4 kv = krow[d4];
        float a = sv.x * kv.x, b = sv.y * kv.y, c = sv.z * kv.z, w = sv.w * kv.w;
        acc = fmaf(a, a, acc); acc = fmaf(b, b, acc);
        acc = fmaf(c, c, acc); acc = fmaf(w, w, acc);
    }
    dst[t] = rsqrtf(fmaxf(acc, 1e-12f));
}

// ---------------- Phase 1: per-(b,p) 200x200x300 GEMM on WMMA + max over j -------------
__global__ __launch_bounds__(256) void mpm_gemm(const float* __restrict__ s1,
                                                const float* __restrict__ s2,
                                                const float* __restrict__ kern,
                                                const float* __restrict__ rsq1,
                                                const float* __restrict__ rsq2,
                                                float* __restrict__ out) {
    const int mblk = blockIdx.x;          // 0..1 -> mtiles mblk*8 .. mblk*8+7
    const int p    = blockIdx.y;          // 0..15
    const int b    = blockIdx.z;          // 0..63
    const int tid  = threadIdx.x;         // 0..255
    const int wave = tid >> 5;            // 0..7
    const int lane = tid & 31;
    const int mtile = mblk * 8 + wave;    // 0..15, valid < 13
    const int half = lane >> 4;
    const int lm   = lane & 15;

    // double-buffered chunk tiles: 2*(208+128)*72 f16 = 96,768 B (<320KB WGP LDS)
    __shared__ alignas(16) _Float16 Bs[2 * NROW * KP];
    __shared__ alignas(16) _Float16 As[2 * MROWS * KP];

    v8f acc[NT];
#pragma unroll
    for (int nt = 0; nt < NT; ++nt)
#pragma unroll
        for (int e = 0; e < 8; ++e) acc[nt][e] = 0.f;

    // loader mapping: lc = float4 column (0..15), lr = row phase (0..15)
    const int lc = tid & 15;
    const int lr = tid >> 4;

    auto fill = [&](int kc) {
        const int buf = kc & 1;
        _Float16* Ab = As + buf * (MROWS * KP);
        _Float16* Bb = Bs + buf * (NROW * KP);
        const int k0 = kc * KC;
        int nv4 = (DD - k0) >> 2;                 // valid float4 columns this chunk
        if (nv4 > 16) nv4 = 16;
        const bool cvalid = (lc < nv4);

        float4 kf = {0.f, 0.f, 0.f, 0.f};
        if (cvalid) kf = *(const float4*)&kern[(size_t)p * DD + k0 + lc * 4];

        // B tile: u2[j,k] = s2*k_p*rsq2, zero-padded (rows>=200 or k>=300)
        for (int r = lr; r < NROW; r += 16) {
            float4 v = {0.f, 0.f, 0.f, 0.f};
            if (r < LL && cvalid) {
                float4 sv = *(const float4*)&s2[((size_t)(b * LL + r)) * DD + k0 + lc * 4];
                float rs  = rsq2[(b * LL + r) * PP + p];
                v.x = sv.x * kf.x * rs; v.y = sv.y * kf.y * rs;
                v.z = sv.z * kf.z * rs; v.w = sv.w * kf.w * rs;
            }
            v4h h; h.x = (_Float16)v.x; h.y = (_Float16)v.y;
            h.z = (_Float16)v.z; h.w = (_Float16)v.w;
            *(v4h*)&Bb[r * KP + lc * 4] = h;
        }
        // A tile: this workgroup's 128 rows
        for (int r = lr; r < MROWS; r += 16) {
            const int i = mblk * MROWS + r;
            float4 v = {0.f, 0.f, 0.f, 0.f};
            if (i < LL && cvalid) {
                float4 sv = *(const float4*)&s1[((size_t)(b * LL + i)) * DD + k0 + lc * 4];
                float rs  = rsq1[(b * LL + i) * PP + p];
                v.x = sv.x * kf.x * rs; v.y = sv.y * kf.y * rs;
                v.z = sv.z * kf.z * rs; v.w = sv.w * kf.w * rs;
            }
            v4h h; h.x = (_Float16)v.x; h.y = (_Float16)v.y;
            h.z = (_Float16)v.z; h.w = (_Float16)v.w;
            *(v4h*)&Ab[r * KP + lc * 4] = h;
        }
    };

    fill(0);
    for (int kc = 0; kc < NKC; ++kc) {
        __syncthreads();                 // fill(kc) visible; compute(kc-1) reads done
        if (kc + 1 < NKC) fill(kc + 1);  // writes buffer (kc+1)&1, disjoint from reads

        if (mtile < NT) {                // wave-uniform: EXEC all-ones for WMMA
            const _Float16* Ab = As + (kc & 1) * (MROWS * KP);
            const _Float16* Bb = Bs + (kc & 1) * (NROW * KP);
#pragma unroll
            for (int kk = 0; kk < 2; ++kk) {
                // A fragment: lane m=lm, contiguous k = half*16 .. half*16+15
                const _Float16* arow = &Ab[(wave * 16 + lm) * KP + kk * 32];
                v8h alo = *(const v8h*)(arow + half * 16);
                v8h ahi = *(const v8h*)(arow + half * 16 + 8);
                v16h a = __builtin_shufflevector(alo, ahi,
                         0,1,2,3,4,5,6,7,8,9,10,11,12,13,14,15);

                // distance-2 software pipeline on B fragments
                const _Float16* br0 = &Bb[lm * KP + kk * 32 + half * 16];
                const _Float16* br1 = &Bb[(16 + lm) * KP + kk * 32 + half * 16];
                v8h b0lo = *(const v8h*)(br0);
                v8h b0hi = *(const v8h*)(br0 + 8);
                v8h b1lo = *(const v8h*)(br1);
                v8h b1hi = *(const v8h*)(br1 + 8);
#pragma unroll
                for (int nt = 0; nt < NT; ++nt) {
                    v8h nlo = b0lo, nhi = b0hi;         // default: keep chain sane
                    if (nt + 2 < NT) {
                        const _Float16* brn =
                            &Bb[((nt + 2) * 16 + lm) * KP + kk * 32 + half * 16];
                        nlo = *(const v8h*)(brn);
                        nhi = *(const v8h*)(brn + 8);
                    }
                    v16h bfrag = __builtin_shufflevector(b0lo, b0hi,
                                 0,1,2,3,4,5,6,7,8,9,10,11,12,13,14,15);
                    acc[nt] = __builtin_amdgcn_wmma_f32_16x16x32_f16(
                        false, a, false, bfrag, (short)0, acc[nt], false, false);
                    b0lo = b1lo; b0hi = b1hi;           // rotate (SSA renames away)
                    b1lo = nlo;  b1hi = nhi;
                }
            }
            // Pin the schedule: burst of DS reads, then (WMMA, 2xDS-read) steady state.
            // Region totals: 56 DS reads (2xA per kk + 26xB per kk), 26 WMMAs.
            __builtin_amdgcn_sched_group_barrier(0x100, 8, 0);   // 8 DS reads up front
#pragma unroll
            for (int it = 0; it < 24; ++it) {
                __builtin_amdgcn_sched_group_barrier(0x008, 1, 0); // 1 WMMA
                __builtin_amdgcn_sched_group_barrier(0x100, 2, 0); // 2 DS reads
            }
            __builtin_amdgcn_sched_group_barrier(0x008, 2, 0);     // drain WMMAs
        }
    }

    // ---- max over j, then over lanes (N lives in lane index of C/D layout) ----
    if (mtile < NT) {
        v8f mx = acc[0];
#pragma unroll
        for (int nt = 1; nt < NT; ++nt) {
            // tile 12 covers j=192..207; j>=200 <=> column lane lm>=8 -> exclude
            if (nt == NT - 1 && lm >= 8) continue;
#pragma unroll
            for (int e = 0; e < 8; ++e) mx[e] = fmaxf(mx[e], acc[nt][e]);
        }
#pragma unroll
        for (int r = 0; r < 8; ++r) {
            float v = mx[r];
            v = fmaxf(v, __shfl_xor(v, 8, 16));
            v = fmaxf(v, __shfl_xor(v, 4, 16));
            v = fmaxf(v, __shfl_xor(v, 2, 16));
            v = fmaxf(v, __shfl_xor(v, 1, 16));
            if (lm == 0) {
                const int i = mtile * 16 + half * 8 + r;   // C/D: VGPR r -> M = r + 8*half
                if (i < LL) out[((size_t)(b * LL) + i) * PP + p] = v;
            }
        }
    }
}

extern "C" void kernel_launch(void* const* d_in, const int* in_sizes, int n_in,
                              void* d_out, int out_size, void* d_ws, size_t ws_size,
                              hipStream_t stream) {
    (void)in_sizes; (void)n_in; (void)out_size; (void)ws_size;
    const float* s1   = (const float*)d_in[0];   // (64,200,300) f32
    const float* s2   = (const float*)d_in[1];   // (64,200,300) f32
    const float* kern = (const float*)d_in[2];   // (16,300)     f32
    float* rsq1 = (float*)d_ws;                  // B*L*P floats
    float* rsq2 = rsq1 + BB * LL * PP;           // B*L*P floats

    const int totalN = 2 * BB * LL * PP;
    mpm_norms<<<(totalN + 255) / 256, 256, 0, stream>>>(s1, s2, kern, rsq1, rsq2);

    dim3 grid(2, PP, BB);                        // (m-block, p, b)
    mpm_gemm<<<grid, 256, 0, stream>>>(s1, s2, kern, rsq1, rsq2, (float*)d_out);
}
